// MMDLoss_75711683493950
// MI455X (gfx1250) — matrix operand order repaired and checked by reading, not compiled, not run on hardware
//
#include <hip/hip_runtime.h>
#include <hip/hip_bf16.h>

// ---------------------------------------------------------------------------
// MMD loss, MI455X (gfx1250): fused bf16x3-split WMMA Gram + exp + reduction
// Pre-converted bf16 hi/lo operands; double-buffered LDS + register prefetch.
// amdgpu_waves_per_eu(1,4): lift VGPR budget so prefetch regs never spill.
// ---------------------------------------------------------------------------

#define NROWS 4096
#define MROWS 4096
#define DDIM  1024
#define SIGMA2_INV (1.0f / (45.0f * 45.0f))

typedef __bf16 bf16_t;
typedef bf16_t v16bf __attribute__((ext_vector_type(16)));
typedef bf16_t v8bf  __attribute__((ext_vector_type(8)));
typedef float  v8f   __attribute__((ext_vector_type(8)));

#define LSTR 40                 // LDS row stride in shorts (32 data + 8 pad, conflict-free)
#define BUFSH (128 * LSTR)      // shorts per tile buffer
#define KSTEPS (DDIM / 32)
// dynamic LDS: 8 tile buffers (2 stages x {Ahi,Alo,Bhi,Blo}) + 2x128 norms + 8 red
#define SMEM_BYTES (8 * BUFSH * 2 + 2 * 128 * 4 + 8 * 4)

// ---- fp32 -> bf16 round-to-nearest-even, and back -------------------------
__device__ __forceinline__ unsigned short bf16_rn(float f) {
    unsigned int u = __float_as_uint(f);
    u += 0x7FFFu + ((u >> 16) & 1u);
    return (unsigned short)(u >> 16);
}
__device__ __forceinline__ float bf16_f(unsigned short h) {
    return __uint_as_float(((unsigned int)h) << 16);
}
__device__ __forceinline__ unsigned long long pack4(unsigned short a, unsigned short b,
                                                    unsigned short c, unsigned short d) {
    return (unsigned long long)a | ((unsigned long long)b << 16) |
           ((unsigned long long)c << 32) | ((unsigned long long)d << 48);
}

// ---- fragment gather from LDS (two 16B chunks -> v16bf) --------------------
// A frag (16x32): lane half=0 -> K {0..7,16..23}; half=1 -> K {8..15,24..31}
// B frag (32x16): lane half=0 -> K {0..7,16..23}; half=1 -> K {8..15,24..31}
__device__ __forceinline__ v16bf frag_from(const unsigned short* __restrict__ base,
                                           int row, int o0, int o1) {
    const v8bf c0 = *(const v8bf*)(base + row * LSTR + o0);
    const v8bf c1 = *(const v8bf*)(base + row * LSTR + o1);
    v16bf r;
#pragma unroll
    for (int i = 0; i < 8; ++i) { r[i] = c0[i]; r[i + 8] = c1[i]; }
    return r;
}

// ---- shared epilogue helper ------------------------------------------------
__device__ __forceinline__ float block_reduce_store(float s, int lane, int wav, float* sRed) {
#pragma unroll
    for (int off = 16; off; off >>= 1) s += __shfl_xor(s, off, 32);
    if (lane == 0) sRed[wav] = s;
    __syncthreads();
    float t = 0.0f;
    if (threadIdx.x == 0) {
#pragma unroll
        for (int i = 0; i < 8; ++i) t += sRed[i];
    }
    return t;
}

// ---- row squared-norms (fallback path) -------------------------------------
__global__ __launch_bounds__(256) void row_norms_kernel(const float* __restrict__ A,
                                                        float* __restrict__ out) {
    __shared__ float red[8];
    const int r = blockIdx.x;
    const float* p = A + (size_t)r * DDIM;
    float s = 0.0f;
    for (int t = threadIdx.x; t < DDIM; t += 256) { float v = p[t]; s += v * v; }
    float t = block_reduce_store(s, threadIdx.x & 31, threadIdx.x >> 5, red);
    if (threadIdx.x == 0) out[r] = t;
}

// ---- one-pass convert fp32 -> bf16 hi/lo + row norm ------------------------
__global__ __launch_bounds__(256) void convert_norm_kernel(const float* __restrict__ src,
                                                           unsigned short* __restrict__ hi,
                                                           unsigned short* __restrict__ lo,
                                                           float* __restrict__ norms) {
    __shared__ float red[8];
    const int r = blockIdx.x;
    const float* p = src + (size_t)r * DDIM;
    unsigned short* ph = hi + (size_t)r * DDIM;
    unsigned short* pl = lo + (size_t)r * DDIM;
    float s = 0.0f;
#pragma unroll
    for (int it = 0; it < DDIM / (256 * 4); ++it) {
        const int t = (threadIdx.x + it * 256) * 4;
        const float4 v = *(const float4*)(p + t);
        s += v.x * v.x + v.y * v.y + v.z * v.z + v.w * v.w;
        unsigned short h0 = bf16_rn(v.x), h1 = bf16_rn(v.y), h2 = bf16_rn(v.z), h3 = bf16_rn(v.w);
        unsigned short l0 = bf16_rn(v.x - bf16_f(h0));
        unsigned short l1 = bf16_rn(v.y - bf16_f(h1));
        unsigned short l2 = bf16_rn(v.z - bf16_f(h2));
        unsigned short l3 = bf16_rn(v.w - bf16_f(h3));
        *(unsigned long long*)(ph + t) = pack4(h0, h1, h2, h3);
        *(unsigned long long*)(pl + t) = pack4(l0, l1, l2, l3);
    }
    float t = block_reduce_store(s, threadIdx.x & 31, threadIdx.x >> 5, red);
    if (threadIdx.x == 0) norms[r] = t;
}

// ===========================================================================
// Main GEMM kernel (pre-converted bf16 hi/lo operands): hot loop is pure
// ds_load + v_wmma; staging is raw b128 copy, pipelined + double-buffered.
// ===========================================================================
__global__ __launch_bounds__(256) __attribute__((amdgpu_waves_per_eu(1, 4)))
void mmd_gemm_pre_kernel(
    const unsigned short* __restrict__ Ahi_g, const unsigned short* __restrict__ Alo_g,
    const unsigned short* __restrict__ Bhi_g, const unsigned short* __restrict__ Blo_g,
    const float* __restrict__ normA, const float* __restrict__ normB,
    double* __restrict__ sumOut, int sym) {
    const int bx = blockIdx.x, by = blockIdx.y;
    if (sym && bx > by) return;

    extern __shared__ char smem[];
    unsigned short* const base = (unsigned short*)smem;
    float* const sNA  = (float*)(base + 8 * BUFSH);
    float* const sNB  = sNA + 128;
    float* const sRed = sNB + 128;

    const int tid  = threadIdx.x;
    const int lane = tid & 31;
    const int wav  = tid >> 5;
    const int wRow = wav >> 2;          // 0..1  (64 rows each)
    const int wCol = wav & 3;           // 0..3  (32 cols each)
    const int ml   = lane & 15;
    const int half = lane >> 4;

    const int RB = by * 128;
    const int CB = bx * 128;

    if (tid < 128)        sNA[tid]       = normA[RB + tid];
    else                  sNB[tid - 128] = normB[CB + (tid - 128)];

    // Staging: 128x32 bf16 tile = 512 x 16B chunks; thread handles chunks
    // q = tid and tid+256 for each of the 4 buffers -> 8 b128 loads/stores.
    const int q0 = tid,        row0 = q0 >> 2, c0 = q0 & 3;
    const int q1 = tid + 256,  row1 = q1 >> 2, c1 = q1 & 3;
    const int lo0 = row0 * LSTR + c0 * 8;
    const int lo1 = row1 * LSTR + c1 * 8;

    uint4 pr0, pr1, pr2, pr3, pr4, pr5, pr6, pr7;   // named regs: SROA-proof
    auto gload = [&](int k0) {
        const size_t oa0 = (size_t)(RB + row0) * DDIM + k0 + c0 * 8;
        const size_t ob0 = (size_t)(CB + row0) * DDIM + k0 + c0 * 8;
        const size_t oa1 = (size_t)(RB + row1) * DDIM + k0 + c1 * 8;
        const size_t ob1 = (size_t)(CB + row1) * DDIM + k0 + c1 * 8;
        pr0 = *(const uint4*)(Ahi_g + oa0);
        pr1 = *(const uint4*)(Alo_g + oa0);
        pr2 = *(const uint4*)(Bhi_g + ob0);
        pr3 = *(const uint4*)(Blo_g + ob0);
        pr4 = *(const uint4*)(Ahi_g + oa1);
        pr5 = *(const uint4*)(Alo_g + oa1);
        pr6 = *(const uint4*)(Bhi_g + ob1);
        pr7 = *(const uint4*)(Blo_g + ob1);
    };
    auto store_stage = [&](int p) {
        unsigned short* const b = base + p * 4 * BUFSH;
        *(uint4*)(b + 0 * BUFSH + lo0) = pr0;
        *(uint4*)(b + 1 * BUFSH + lo0) = pr1;
        *(uint4*)(b + 2 * BUFSH + lo0) = pr2;
        *(uint4*)(b + 3 * BUFSH + lo0) = pr3;
        *(uint4*)(b + 0 * BUFSH + lo1) = pr4;
        *(uint4*)(b + 1 * BUFSH + lo1) = pr5;
        *(uint4*)(b + 2 * BUFSH + lo1) = pr6;
        *(uint4*)(b + 3 * BUFSH + lo1) = pr7;
    };

    v8f acc[4][2];
#pragma unroll
    for (int mi = 0; mi < 4; ++mi)
#pragma unroll
        for (int ni = 0; ni < 2; ++ni)
#pragma unroll
            for (int e = 0; e < 8; ++e) acc[mi][ni][e] = 0.0f;

    auto compute_step = [&](int p) {
        const unsigned short* Ahi = base + (p * 4 + 0) * BUFSH;
        const unsigned short* Alo = base + (p * 4 + 1) * BUFSH;
        const unsigned short* Bhi = base + (p * 4 + 2) * BUFSH;
        const unsigned short* Blo = base + (p * 4 + 3) * BUFSH;
        v16bf bh[2], bl[2];
#pragma unroll
        for (int ni = 0; ni < 2; ++ni) {
            const int brow = wCol * 32 + ni * 16 + ml;
            bh[ni] = frag_from(Bhi, brow, 8 * half, 16 + 8 * half);
            bl[ni] = frag_from(Blo, brow, 8 * half, 16 + 8 * half);
        }
#pragma unroll
        for (int mi = 0; mi < 4; ++mi) {
            const int arow = wRow * 64 + mi * 16 + ml;
            const v16bf ah = frag_from(Ahi, arow, 8 * half, 16 + 8 * half);
            const v16bf al = frag_from(Alo, arow, 8 * half, 16 + 8 * half);
#pragma unroll
            for (int ni = 0; ni < 2; ++ni) {
                acc[mi][ni] = __builtin_amdgcn_wmma_f32_16x16x32_bf16(
                    false, ah, false, bh[ni], (short)0, acc[mi][ni], false, false);
                acc[mi][ni] = __builtin_amdgcn_wmma_f32_16x16x32_bf16(
                    false, ah, false, bl[ni], (short)0, acc[mi][ni], false, false);
                acc[mi][ni] = __builtin_amdgcn_wmma_f32_16x16x32_bf16(
                    false, al, false, bh[ni], (short)0, acc[mi][ni], false, false);
            }
        }
    };

    gload(0);
    store_stage(0);
    __syncthreads();

    int p = 0;
#pragma unroll 1
    for (int kt = 0; kt < KSTEPS; ++kt) {
        const bool more = (kt + 1) < KSTEPS;
        if (more) gload((kt + 1) * 32);   // latency hides under WMMAs
        compute_step(p);                  // 24 x v_wmma
        if (more) store_stage(p ^ 1);
        __syncthreads();
        p ^= 1;
    }

    float s = 0.0f;
#pragma unroll
    for (int mi = 0; mi < 4; ++mi) {
        const int lmBase = wRow * 64 + mi * 16 + 8 * half;
#pragma unroll
        for (int ni = 0; ni < 2; ++ni) {
            const int ln  = wCol * 32 + ni * 16 + ml;
            const float yn = sNB[ln];
            const int   gn = CB + ln;
#pragma unroll
            for (int e = 0; e < 8; ++e) {
                const int lm = lmBase + e;
                const float d2 = sNA[lm] + yn - 2.0f * acc[mi][ni][e];
                const float v  = __expf(-d2 * SIGMA2_INV);
                if (!(sym && (RB + lm) == gn)) s += v;
            }
        }
    }
    if (sym && bx != by) s *= 2.0f;

    float t = block_reduce_store(s, lane, wav, sRed);
    if (tid == 0) atomicAdd(sumOut, (double)t);
}

// ===========================================================================
// Fallback GEMM kernel (converts fp32 -> bf16 hi/lo inside the loop);
// used only if ws_size cannot hold the pre-converted operands.
// ===========================================================================
__global__ __launch_bounds__(256) __attribute__((amdgpu_waves_per_eu(1, 4)))
void mmd_gemm_fused_kernel(
    const float* __restrict__ Amat, const float* __restrict__ Bmat,
    const float* __restrict__ normA, const float* __restrict__ normB,
    double* __restrict__ sumOut, int sym) {
    const int bx = blockIdx.x, by = blockIdx.y;
    if (sym && bx > by) return;

    extern __shared__ char smem[];
    unsigned short* const base = (unsigned short*)smem;
    float* const sNA  = (float*)(base + 8 * BUFSH);
    float* const sNB  = sNA + 128;
    float* const sRed = sNB + 128;

    const int tid  = threadIdx.x;
    const int lane = tid & 31;
    const int wav  = tid >> 5;
    const int wRow = wav >> 2;
    const int wCol = wav & 3;
    const int ml   = lane & 15;
    const int half = lane >> 4;

    const int RB = by * 128;
    const int CB = bx * 128;

    if (tid < 128)        sNA[tid]       = normA[RB + tid];
    else                  sNB[tid - 128] = normB[CB + (tid - 128)];

    float4 pa0, pa1, pa2, pa3, pb0, pb1, pb2, pb3;  // named regs: SROA-proof
    auto gload = [&](int k0) {
        const int r0 = tid >> 3,        cq0 = tid & 7;
        const int r1 = (tid + 256) >> 3, cq1 = tid & 7;
        const int r2 = (tid + 512) >> 3, cq2 = tid & 7;
        const int r3 = (tid + 768) >> 3, cq3 = tid & 7;
        pa0 = *(const float4*)(Amat + (size_t)(RB + r0) * DDIM + k0 + cq0 * 4);
        pa1 = *(const float4*)(Amat + (size_t)(RB + r1) * DDIM + k0 + cq1 * 4);
        pa2 = *(const float4*)(Amat + (size_t)(RB + r2) * DDIM + k0 + cq2 * 4);
        pa3 = *(const float4*)(Amat + (size_t)(RB + r3) * DDIM + k0 + cq3 * 4);
        pb0 = *(const float4*)(Bmat + (size_t)(CB + r0) * DDIM + k0 + cq0 * 4);
        pb1 = *(const float4*)(Bmat + (size_t)(CB + r1) * DDIM + k0 + cq1 * 4);
        pb2 = *(const float4*)(Bmat + (size_t)(CB + r2) * DDIM + k0 + cq2 * 4);
        pb3 = *(const float4*)(Bmat + (size_t)(CB + r3) * DDIM + k0 + cq3 * 4);
    };
    auto cvt1 = [&](unsigned short* hiB, unsigned short* loB, float4 v, int q) {
        const int row = q >> 3, cq = q & 7;
        unsigned short h0 = bf16_rn(v.x), h1 = bf16_rn(v.y), h2 = bf16_rn(v.z), h3 = bf16_rn(v.w);
        unsigned short l0 = bf16_rn(v.x - bf16_f(h0));
        unsigned short l1 = bf16_rn(v.y - bf16_f(h1));
        unsigned short l2 = bf16_rn(v.z - bf16_f(h2));
        unsigned short l3 = bf16_rn(v.w - bf16_f(h3));
        *(unsigned long long*)(hiB + row * LSTR + cq * 4) = pack4(h0, h1, h2, h3);
        *(unsigned long long*)(loB + row * LSTR + cq * 4) = pack4(l0, l1, l2, l3);
    };
    auto store_stage = [&](int p) {
        unsigned short* const ah = base + (p * 4 + 0) * BUFSH;
        unsigned short* const al = base + (p * 4 + 1) * BUFSH;
        unsigned short* const bh = base + (p * 4 + 2) * BUFSH;
        unsigned short* const bl = base + (p * 4 + 3) * BUFSH;
        cvt1(ah, al, pa0, tid);       cvt1(ah, al, pa1, tid + 256);
        cvt1(ah, al, pa2, tid + 512); cvt1(ah, al, pa3, tid + 768);
        cvt1(bh, bl, pb0, tid);       cvt1(bh, bl, pb1, tid + 256);
        cvt1(bh, bl, pb2, tid + 512); cvt1(bh, bl, pb3, tid + 768);
    };

    v8f acc[4][2];
#pragma unroll
    for (int mi = 0; mi < 4; ++mi)
#pragma unroll
        for (int ni = 0; ni < 2; ++ni)
#pragma unroll
            for (int e = 0; e < 8; ++e) acc[mi][ni][e] = 0.0f;

    auto compute_step = [&](int p) {
        const unsigned short* Ahi = base + (p * 4 + 0) * BUFSH;
        const unsigned short* Alo = base + (p * 4 + 1) * BUFSH;
        const unsigned short* Bhi = base + (p * 4 + 2) * BUFSH;
        const unsigned short* Blo = base + (p * 4 + 3) * BUFSH;
        v16bf bh[2], bl[2];
#pragma unroll
        for (int ni = 0; ni < 2; ++ni) {
            const int brow = wCol * 32 + ni * 16 + ml;
            bh[ni] = frag_from(Bhi, brow, 8 * half, 16 + 8 * half);
            bl[ni] = frag_from(Blo, brow, 8 * half, 16 + 8 * half);
        }
#pragma unroll
        for (int mi = 0; mi < 4; ++mi) {
            const int arow = wRow * 64 + mi * 16 + ml;
            const v16bf ah = frag_from(Ahi, arow, 8 * half, 16 + 8 * half);
            const v16bf al = frag_from(Alo, arow, 8 * half, 16 + 8 * half);
#pragma unroll
            for (int ni = 0; ni < 2; ++ni) {
                acc[mi][ni] = __builtin_amdgcn_wmma_f32_16x16x32_bf16(
                    false, ah, false, bh[ni], (short)0, acc[mi][ni], false, false);
                acc[mi][ni] = __builtin_amdgcn_wmma_f32_16x16x32_bf16(
                    false, ah, false, bl[ni], (short)0, acc[mi][ni], false, false);
                acc[mi][ni] = __builtin_amdgcn_wmma_f32_16x16x32_bf16(
                    false, al, false, bh[ni], (short)0, acc[mi][ni], false, false);
            }
        }
    };

    gload(0);
    store_stage(0);
    __syncthreads();

    int p = 0;
#pragma unroll 1
    for (int kt = 0; kt < KSTEPS; ++kt) {
        const bool more = (kt + 1) < KSTEPS;
        if (more) gload((kt + 1) * 32);
        compute_step(p);
        if (more) store_stage(p ^ 1);
        __syncthreads();
        p ^= 1;
    }

    float s = 0.0f;
#pragma unroll
    for (int mi = 0; mi < 4; ++mi) {
        const int lmBase = wRow * 64 + mi * 16 + 8 * half;
#pragma unroll
        for (int ni = 0; ni < 2; ++ni) {
            const int ln  = wCol * 32 + ni * 16 + ml;
            const float yn = sNB[ln];
            const int   gn = CB + ln;
#pragma unroll
            for (int e = 0; e < 8; ++e) {
                const int lm = lmBase + e;
                const float d2 = sNA[lm] + yn - 2.0f * acc[mi][ni][e];
                const float v  = __expf(-d2 * SIGMA2_INV);
                if (!(sym && (RB + lm) == gn)) s += v;
            }
        }
    }
    if (sym && bx != by) s *= 2.0f;

    float t = block_reduce_store(s, lane, wav, sRed);
    if (tid == 0) atomicAdd(sumOut, (double)t);
}

__global__ void init_sums_kernel(double* s) {
    if (threadIdx.x < 3) s[threadIdx.x] = 0.0;
}

__global__ void finalize_kernel(const double* __restrict__ s, float* __restrict__ out) {
    if (threadIdx.x == 0) {
        const double n = (double)NROWS, m = (double)MROWS;
        double mmd = s[0] / (n * (n - 1.0))
                   + s[1] / (m * (m - 1.0))
                   - (2.0 / (n * m)) * s[2];
        out[0] = (float)mmd;
    }
}

extern "C" void kernel_launch(void* const* d_in, const int* in_sizes, int n_in,
                              void* d_out, int out_size, void* d_ws, size_t ws_size,
                              hipStream_t stream) {
    const float* X = (const float*)d_in[0];   // inputs  [4096,1024] f32
    const float* Y = (const float*)d_in[1];   // samples [4096,1024] f32
    float* out = (float*)d_out;

    // workspace layout
    char* ws = (char*)d_ws;
    double* sums = (double*)ws;                                   // 3 doubles
    float*  normX = (float*)(ws + 256);                           // 16 KB
    float*  normY = (float*)(ws + 256 + NROWS * 4);               // 16 KB
    const size_t MATB = (size_t)NROWS * DDIM * 2;                 // 8 MB per bf16 matrix
    char*   conv = ws + 256 + (NROWS + MROWS) * 4;                // 64B-aligned
    unsigned short* Xhi = (unsigned short*)(conv + 0 * MATB);
    unsigned short* Xlo = (unsigned short*)(conv + 1 * MATB);
    unsigned short* Yhi = (unsigned short*)(conv + 2 * MATB);
    unsigned short* Ylo = (unsigned short*)(conv + 3 * MATB);
    const size_t need = (size_t)(conv - ws) + 4 * MATB;

    init_sums_kernel<<<1, 32, 0, stream>>>(sums);

    dim3 grid(MROWS / 128, NROWS / 128);
    dim3 block(256);

    if (ws_size >= need) {
        // fast path: pre-convert to bf16 hi/lo (fused with row norms)
        convert_norm_kernel<<<NROWS, 256, 0, stream>>>(X, Xhi, Xlo, normX);
        convert_norm_kernel<<<MROWS, 256, 0, stream>>>(Y, Yhi, Ylo, normY);
        mmd_gemm_pre_kernel<<<grid, block, SMEM_BYTES, stream>>>(
            Xhi, Xlo, Xhi, Xlo, normX, normX, &sums[0], 1);
        mmd_gemm_pre_kernel<<<grid, block, SMEM_BYTES, stream>>>(
            Yhi, Ylo, Yhi, Ylo, normY, normY, &sums[1], 1);
        mmd_gemm_pre_kernel<<<grid, block, SMEM_BYTES, stream>>>(
            Xhi, Xlo, Yhi, Ylo, normX, normY, &sums[2], 0);
    } else {
        // fallback: convert inside the GEMM loop
        row_norms_kernel<<<NROWS, 256, 0, stream>>>(X, normX);
        row_norms_kernel<<<MROWS, 256, 0, stream>>>(Y, normY);
        mmd_gemm_fused_kernel<<<grid, block, SMEM_BYTES, stream>>>(
            X, X, normX, normX, &sums[0], 1);
        mmd_gemm_fused_kernel<<<grid, block, SMEM_BYTES, stream>>>(
            Y, Y, normY, normY, &sums[1], 1);
        mmd_gemm_fused_kernel<<<grid, block, SMEM_BYTES, stream>>>(
            X, Y, normX, normY, &sums[2], 0);
    }

    finalize_kernel<<<1, 32, 0, stream>>>(sums, out);
}
